// WindowAttention_41875931136532
// MI455X (gfx1250) — matrix operand order repaired and checked by reading, not compiled, not run on hardware
//
#include <hip/hip_runtime.h>
#include <hip/hip_bf16.h>

#define WIN   7
#define HEADS 6
#define DIM   192
#define NTOK  49
#define NPAD  64
#define DH    32
#define THREADS 384   // 12 waves (wave32)

typedef __attribute__((ext_vector_type(16))) _Float16 v16h;
typedef __attribute__((ext_vector_type(8)))  _Float16 v8h;
typedef __attribute__((ext_vector_type(8)))  float    v8f;

union FragU { v16h v; v8h h[2]; };

__device__ __forceinline__ v8f vzero() {
    v8f z = {0.f,0.f,0.f,0.f,0.f,0.f,0.f,0.f};
    return z;
}

// A fragment (16x32 f16, M x K): row-major source, stride in halves.
// Lane layout (ISA 7.12.2): lanes 0-15 M=0-15 (hi=0), lanes 16-31 M=0-15 (hi=1);
// elements 0..7 -> K = hi*8 + e, elements 8..15 -> K = 16 + hi*8 + (e-8).
__device__ __forceinline__ v16h load_a_frag(const _Float16* p, int stride,
                                            int r0, int c0, int lane) {
    const int m  = lane & 15;
    const int hi = lane >> 4;
    const _Float16* q = p + (r0 + m) * stride + c0 + hi * 8;
    FragU f;
    f.h[0] = *(const v8h*)(q);        // K = hi*8 + 0..7
    f.h[1] = *(const v8h*)(q + 16);   // K = 16 + hi*8 + 0..7
    return f.v;
}

// B fragment (32x16 f16, K x N): source laid out [N][K] row-major (K contiguous).
// Lane layout: N = lane&15; elements 0..7 -> K = hi*8 + e, elements 8..15 -> K = 16 + hi*8 + (e-8).
__device__ __forceinline__ v16h load_b_frag(const _Float16* p, int stride,
                                            int n0, int k0, int lane) {
    const int n  = lane & 15;
    const int hi = lane >> 4;
    const _Float16* q = p + (n0 + n) * stride + k0 + hi * 8;
    FragU f;
    f.h[0] = *(const v8h*)(q);
    f.h[1] = *(const v8h*)(q + 16);
    return f.v;
}

__device__ __forceinline__ v8f wmma_f16(v16h a, v16h b, v8f c) {
    return __builtin_amdgcn_wmma_f32_16x16x32_f16(false, a, false, b,
                                                  (short)0, c, false, false);
}

// ---------------------------------------------------------------------------
// Prep: f32->f16 weight copies + padded per-head bias matrix [6][64][64]
// ---------------------------------------------------------------------------
__global__ void prep_kernel(const float* __restrict__ qkv_w,
                            const float* __restrict__ proj_w,
                            const float* __restrict__ bias_table,
                            const int*   __restrict__ rel_idx,
                            _Float16* __restrict__ qkvw_h,
                            _Float16* __restrict__ projw_h,
                            float* __restrict__ bias_pad) {
    int i = blockIdx.x * blockDim.x + threadIdx.x;
    if (i < 3 * DIM * DIM) qkvw_h[i] = (_Float16)qkv_w[i];
    if (i < DIM * DIM)     projw_h[i] = (_Float16)proj_w[i];
    if (i < HEADS * NPAD * NPAD) {
        int h = i / (NPAD * NPAD);
        int r = i % (NPAD * NPAD);
        int n = r / NPAD, m = r % NPAD;
        float v = 0.0f;
        if (n < NTOK) {
            v = (m < NTOK) ? bias_table[rel_idx[n * NTOK + m] * HEADS + h]
                           : -30000.0f;   // kill padded key columns in softmax
        }
        bias_pad[i] = v;
    }
}

// ---------------------------------------------------------------------------
// Fused window attention: one workgroup per window.
// LDS (96 KB): [0)      xbuf   [64][192] f16  (later reused as attn-out)
//              [24576B) qkbase per head 8KB: q[64][32] + k[64][32] -> probs[64][64]
//              [73728B) vtbase per head 4KB: vT[32][64]
// ---------------------------------------------------------------------------
__global__ __launch_bounds__(THREADS)
void win_attn_kernel(const float* __restrict__ x,
                     const float* __restrict__ qkv_b,
                     const float* __restrict__ proj_b,
                     const _Float16* __restrict__ qkvw_h,
                     const _Float16* __restrict__ projw_h,
                     const float* __restrict__ bias_pad,
                     float* __restrict__ out) {
    __shared__ _Float16 smem[49152];    // 96 KB
    const int tid  = threadIdx.x;
    const int w    = tid >> 5;          // wave id 0..11
    const int lane = tid & 31;
    const int hi   = lane >> 4;
    const int ln   = lane & 15;
    const int blk  = blockIdx.x;

    _Float16* xbuf   = smem;            // [64][192]
    _Float16* qkbase = smem + 12288;    // + h*4096 halves
    _Float16* vtbase = smem + 36864;    // + h*2048 halves

    // ---- Phase 1: stage x window (f32 -> f16), zero-pad rows 49..63 ----
    #pragma unroll 4
    for (int i = tid; i < NPAD * DIM; i += THREADS) {
        int row = i / DIM, col = i % DIM;
        float v = (row < NTOK) ? x[(blk * NTOK + row) * DIM + col] : 0.0f;
        xbuf[i] = (_Float16)v;
    }
    __syncthreads();

    // ---- Phase 2: QKV projection, M=64 N=576 K=192 ----
    // wave w owns output cols [w*48, w*48+48): 4 M-tiles x 3 N-tiles
    {
        v8f acc[4][3];
        #pragma unroll
        for (int mt = 0; mt < 4; ++mt)
            #pragma unroll
            for (int nt = 0; nt < 3; ++nt) acc[mt][nt] = vzero();

        #pragma unroll
        for (int ks = 0; ks < 6; ++ks) {
            const int k0 = ks * 32;
            v16h a[4], b[3];
            #pragma unroll
            for (int mt = 0; mt < 4; ++mt)
                a[mt] = load_a_frag(xbuf, DIM, mt * 16, k0, lane);
            #pragma unroll
            for (int nt = 0; nt < 3; ++nt)
                b[nt] = load_b_frag(qkvw_h, DIM, w * 48 + nt * 16, k0, lane);
            #pragma unroll
            for (int mt = 0; mt < 4; ++mt)
                #pragma unroll
                for (int nt = 0; nt < 3; ++nt)
                    acc[mt][nt] = wmma_f16(a[mt], b[nt], acc[mt][nt]);
        }

        const int s = w >> 2;                      // 0=q 1=k 2=v (uniform per wave)
        const float scale = 0.17677669529663687f;  // Dh^-0.5
        #pragma unroll
        for (int nt = 0; nt < 3; ++nt) {
            const int j = w * 48 + nt * 16 + ln;   // global qkv output column
            const float bias = qkv_b[j];
            const int h = (j >> 5) % HEADS;
            const int d = j & 31;
            #pragma unroll
            for (int mt = 0; mt < 4; ++mt) {
                #pragma unroll
                for (int r = 0; r < 8; ++r) {
                    const int row = mt * 16 + hi * 8 + r;
                    const float v = acc[mt][nt][r] + bias;
                    if (s == 0)
                        qkbase[h * 4096 + row * DH + d] = (_Float16)(v * scale);
                    else if (s == 1)
                        qkbase[h * 4096 + 2048 + row * DH + d] = (_Float16)v;
                    else
                        vtbase[h * 2048 + d * NPAD + row] = (_Float16)v;  // V^T
                }
            }
        }
    }
    __syncthreads();

    // ---- Phase 3: scores + softmax (2 waves per head) ----
    const int h   = w >> 1;
    const int sub = w & 1;       // M rows [sub*32, sub*32+32)
    v8f pr[2][4];
    {
        const _Float16* qh = qkbase + h * 4096;
        const _Float16* kh = qh + 2048;
        #pragma unroll
        for (int mtl = 0; mtl < 2; ++mtl)
            #pragma unroll
            for (int nt = 0; nt < 4; ++nt) pr[mtl][nt] = vzero();

        v16h aq[2], bk[4];
        #pragma unroll
        for (int mtl = 0; mtl < 2; ++mtl)
            aq[mtl] = load_a_frag(qh, DH, (sub * 2 + mtl) * 16, 0, lane);
        #pragma unroll
        for (int nt = 0; nt < 4; ++nt)
            bk[nt] = load_b_frag(kh, DH, nt * 16, 0, lane);   // K row-major == B^T layout
        #pragma unroll
        for (int mtl = 0; mtl < 2; ++mtl)
            #pragma unroll
            for (int nt = 0; nt < 4; ++nt)
                pr[mtl][nt] = wmma_f16(aq[mtl], bk[nt], pr[mtl][nt]);

        // + relative position bias
        #pragma unroll
        for (int mtl = 0; mtl < 2; ++mtl)
            #pragma unroll
            for (int nt = 0; nt < 4; ++nt)
                #pragma unroll
                for (int r = 0; r < 8; ++r) {
                    const int row = (sub * 2 + mtl) * 16 + hi * 8 + r;
                    const int col = nt * 16 + ln;
                    pr[mtl][nt][r] += bias_pad[h * 4096 + row * NPAD + col];
                }

        // row softmax: row lives across lanes (ln 0..15) x 4 N-tiles for fixed (hi,r)
        #pragma unroll
        for (int mtl = 0; mtl < 2; ++mtl) {
            #pragma unroll
            for (int r = 0; r < 8; ++r) {
                float mx = pr[mtl][0][r];
                #pragma unroll
                for (int nt = 1; nt < 4; ++nt) mx = fmaxf(mx, pr[mtl][nt][r]);
                #pragma unroll
                for (int msk = 1; msk < 16; msk <<= 1)
                    mx = fmaxf(mx, __shfl_xor(mx, msk, 32));
                float sum = 0.f;
                #pragma unroll
                for (int nt = 0; nt < 4; ++nt) {
                    float e = __expf(pr[mtl][nt][r] - mx);
                    pr[mtl][nt][r] = e;
                    sum += e;
                }
                #pragma unroll
                for (int msk = 1; msk < 16; msk <<= 1)
                    sum += __shfl_xor(sum, msk, 32);
                const float inv = 1.0f / sum;
                #pragma unroll
                for (int nt = 0; nt < 4; ++nt) pr[mtl][nt][r] *= inv;
            }
        }
    }
    __syncthreads();   // all waves done reading q/k before probs overwrite them

    {
        _Float16* ph = qkbase + h * 4096;   // probs [64][64], reuses q+k space
        #pragma unroll
        for (int mtl = 0; mtl < 2; ++mtl)
            #pragma unroll
            for (int nt = 0; nt < 4; ++nt)
                #pragma unroll
                for (int r = 0; r < 8; ++r) {
                    const int row = (sub * 2 + mtl) * 16 + hi * 8 + r;
                    const int col = nt * 16 + ln;
                    ph[row * NPAD + col] = (_Float16)pr[mtl][nt][r];
                }
    }
    __syncthreads();

    // ---- Phase 4: O = P @ V  (M=64, K=64, N=32), write attn-out over xbuf ----
    {
        const _Float16* ph = qkbase + h * 4096;
        const _Float16* vt = vtbase + h * 2048;
        v8f oa[2][2];
        #pragma unroll
        for (int mtl = 0; mtl < 2; ++mtl)
            #pragma unroll
            for (int nd = 0; nd < 2; ++nd) oa[mtl][nd] = vzero();

        #pragma unroll
        for (int kk = 0; kk < 2; ++kk) {
            const int k0 = kk * 32;
            v16h ap[2], bv[2];
            #pragma unroll
            for (int mtl = 0; mtl < 2; ++mtl)
                ap[mtl] = load_a_frag(ph, NPAD, (sub * 2 + mtl) * 16, k0, lane);
            #pragma unroll
            for (int nd = 0; nd < 2; ++nd)
                bv[nd] = load_b_frag(vt, NPAD, nd * 16, k0, lane);  // vT: [d][tok]
            #pragma unroll
            for (int mtl = 0; mtl < 2; ++mtl)
                #pragma unroll
                for (int nd = 0; nd < 2; ++nd)
                    oa[mtl][nd] = wmma_f16(ap[mtl], bv[nd], oa[mtl][nd]);
        }
        #pragma unroll
        for (int mtl = 0; mtl < 2; ++mtl)
            #pragma unroll
            for (int nd = 0; nd < 2; ++nd)
                #pragma unroll
                for (int r = 0; r < 8; ++r) {
                    const int row = (sub * 2 + mtl) * 16 + hi * 8 + r;
                    const int col = h * DH + nd * 16 + ln;
                    xbuf[row * DIM + col] = (_Float16)oa[mtl][nd][r];
                }
    }
    __syncthreads();

    // ---- Phase 5: output projection, M=64 N=192 K=192; wave w -> N-tile w ----
    {
        v8f pa[4];
        #pragma unroll
        for (int mt = 0; mt < 4; ++mt) pa[mt] = vzero();

        #pragma unroll
        for (int ks = 0; ks < 6; ++ks) {
            const int k0 = ks * 32;
            v16h a[4];
            #pragma unroll
            for (int mt = 0; mt < 4; ++mt)
                a[mt] = load_a_frag(xbuf, DIM, mt * 16, k0, lane);
            v16h bw = load_b_frag(projw_h, DIM, w * 16, k0, lane);
            #pragma unroll
            for (int mt = 0; mt < 4; ++mt)
                pa[mt] = wmma_f16(a[mt], bw, pa[mt]);
        }

        const int j = w * 16 + ln;
        const float pb = proj_b[j];
        #pragma unroll
        for (int mt = 0; mt < 4; ++mt) {
            #pragma unroll
            for (int r = 0; r < 8; ++r) {
                const int row = mt * 16 + hi * 8 + r;
                if (row < NTOK)
                    out[(blk * NTOK + row) * DIM + j] = pa[mt][r] + pb;
            }
        }
    }
}

// ---------------------------------------------------------------------------
extern "C" void kernel_launch(void* const* d_in, const int* in_sizes, int n_in,
                              void* d_out, int out_size, void* d_ws, size_t ws_size,
                              hipStream_t stream) {
    (void)in_sizes; (void)n_in; (void)out_size; (void)ws_size;
    const float* x          = (const float*)d_in[0];
    const float* qkv_w      = (const float*)d_in[1];
    const float* qkv_b      = (const float*)d_in[2];
    const float* proj_w     = (const float*)d_in[3];
    const float* proj_b     = (const float*)d_in[4];
    const float* bias_table = (const float*)d_in[5];
    const int*   rel_idx    = (const int*)d_in[6];
    float* out = (float*)d_out;

    char* ws = (char*)d_ws;
    float*    bias_pad = (float*)ws;                         // 98304 B
    _Float16* qkvw_h   = (_Float16*)(ws + 98304);            // 221184 B
    _Float16* projw_h  = (_Float16*)(ws + 98304 + 221184);   // 73728 B

    prep_kernel<<<(3 * DIM * DIM + 255) / 256, 256, 0, stream>>>(
        qkv_w, proj_w, bias_table, rel_idx, qkvw_h, projw_h, bias_pad);

    win_attn_kernel<<<4096, THREADS, 0, stream>>>(
        x, qkv_b, proj_b, qkvw_h, projw_h, bias_pad, out);
}
